// Block_86096914416145
// MI455X (gfx1250) — matrix operand order repaired and checked by reading, not compiled, not run on hardware
//
#include <hip/hip_runtime.h>
#include <stdint.h>

// SNN membrane scan + surrogate spike + triangular double-cumsum, MI455X (gfx1250).
//
// Bandwidth-bound: 256 MB mandatory HBM traffic -> ~11us floor @ 23.3 TB/s.
// Strategy: chunk-parallel scan decomposition (32 chunks x 32 steps) so the
// heavy passes have 512K threads; spikes carried between passes as packed
// bits (2 MB); final z/gz pass computes the intra-block double cumsum with
// v_wmma_f32_16x16x32_f16 (exact: integer values, f32 accumulate).
// `current` is read twice but fits in the 192 MB L2 (64 MB, RT policy);
// v_th uses non-temporal loads, outputs use non-temporal stores.

typedef __attribute__((ext_vector_type(16))) _Float16 v16h;
typedef __attribute__((ext_vector_type(8)))  float    v8f;
typedef __attribute__((ext_vector_type(8)))  uint32_t v8u;

#define T_LEN  1024
#define N_IN   512
#define BATCH  32
#define BN     16384                 // BATCH * N_IN
#define NCHUNK 32
#define LCHUNK 32                    // NCHUNK * LCHUNK == T_LEN
#define CBN    (NCHUNK * BN)         // 524288

// ---------- Pass 1: chunk-local membrane endpoint (zero initial state) ----
__global__ void pass1_mend(const float* __restrict__ cur,
                           const float* __restrict__ beta,
                           float* __restrict__ mend) {
  int id = blockIdx.x * blockDim.x + threadIdx.x;     // [0, CBN)
  if (id >= CBN) return;
  int c   = id >> 14;                                  // id / BN
  int col = id & (BN - 1);
  float b = beta[col & (N_IN - 1)];
  const float* p = cur + (size_t)c * LCHUNK * BN + col;
  float m = 0.f;
#pragma unroll
  for (int t = 0; t < LCHUNK; ++t)
    m = fmaf(b, m, p[(size_t)t * BN]);
  mend[id] = m;
}

// ---------- Pass 2: combine membrane carries across chunks ----------------
__global__ void pass2_mcombine(const float* __restrict__ beta,
                               const float* __restrict__ v_init,
                               const float* __restrict__ mend,
                               float* __restrict__ m_in,
                               float* __restrict__ m_last_out) {
  int col = blockIdx.x * blockDim.x + threadIdx.x;     // [0, BN)
  if (col >= BN) return;
  float b  = beta[col & (N_IN - 1)];
  float b2 = b * b, b4 = b2 * b2, b8 = b4 * b4, b16 = b8 * b8;
  float bl = b16 * b16;                                // beta^32
  float m  = v_init[col];
#pragma unroll 4
  for (int c = 0; c < NCHUNK; ++c) {
    size_t i = (size_t)c * BN + col;
    m_in[i] = m;                                       // carry INTO chunk c
    m = fmaf(bl, m, mend[i]);
  }
  m_last_out[col] = m;                                 // output #3
}

// ---------- Pass 3: spikes (packed bits) + per-chunk cumsum stats ---------
__global__ void pass3_spikes(const float* __restrict__ cur,
                             const float* __restrict__ vth,
                             const float* __restrict__ beta,
                             const float* __restrict__ m_in,
                             uint32_t* __restrict__ spk,
                             float* __restrict__ Sc,
                             float* __restrict__ Zc) {
  int id = blockIdx.x * blockDim.x + threadIdx.x;      // [0, CBN)
  if (id >= CBN) return;
  int c   = id >> 14;
  int col = id & (BN - 1);
  float b = beta[col & (N_IN - 1)];
  const float* pc = cur + (size_t)c * LCHUNK * BN + col;
  const float* pv = vth + (size_t)c * LCHUNK * BN + col;
  float m = m_in[id];
  uint32_t bits = 0;
  int s1 = 0, z = 0;
#pragma unroll
  for (int t = 0; t < LCHUNK; ++t) {
    m = fmaf(b, m, pc[(size_t)t * BN]);
    float th = __builtin_nontemporal_load(&pv[(size_t)t * BN]); // read-once: NT
    uint32_t sp = (m >= th) ? 1u : 0u;
    bits |= sp << t;
    s1 += (int)sp;
    z  += s1;
  }
  spk[id] = bits;
  Sc[id] = (float)s1;                                  // spike count in chunk
  Zc[id] = (float)z;                                   // local double-cumsum end
}

// ---------- Pass 4: combine (s1, z) carries across chunks (exact ints) ----
__global__ void pass4_szcombine(const float* __restrict__ Sc,
                                const float* __restrict__ Zc,
                                float* __restrict__ s1_in,
                                float* __restrict__ z_in) {
  int col = blockIdx.x * blockDim.x + threadIdx.x;     // [0, BN)
  if (col >= BN) return;
  float s1 = 0.f, z = 0.f;
#pragma unroll 4
  for (int c = 0; c < NCHUNK; ++c) {
    size_t i = (size_t)c * BN + col;
    s1_in[i] = s1;
    z_in[i]  = z;
    z  += (float)LCHUNK * s1 + Zc[i];
    s1 += Sc[i];
  }
}

// Expand 16 spike bits into the B-matrix v16h (f16 0.0/1.0) with integer
// arithmetic: bit(2w)->half0, bit(2w+1)->half1 of VGPR word w.
// comb = m | (m<<15) places bit 2w+1 at position 2w+16, so each word is
// ((comb >> 2w) & 0x00010001) * 0x3C00  (f16 1.0 = 0x3C00; no carry).
__device__ __forceinline__ v16h expand_bits_f16(uint32_t m) {
  uint32_t comb = m | (m << 15);
  v8u w;
#pragma unroll
  for (int i = 0; i < 8; ++i)
    w[i] = ((comb >> (2 * i)) & 0x00010001u) * 0x3C00u;
  return __builtin_bit_cast(v16h, w);
}

// ---------- Pass 5: z/gz via WMMA triangular matmul -----------------------
// One wave per (chunk, 32-column tile). z[t0+i][n] = z_in + (i+1)*s1_in +
// sum_{j<=i} (i-j+1)*spike[j][n]; the last term is D = A(16x32,f16) x
// B(32x16,f16) on v_wmma_f32_16x16x32_f16 with K>=16 zero-padded.
__global__ void __launch_bounds__(256)
pass5_wmma_out(const uint32_t* __restrict__ spk,
               const float* __restrict__ s1_in,
               const float* __restrict__ z_in,
               float* __restrict__ gz_out,
               float* __restrict__ z_out) {
  const int  lane = threadIdx.x & 31;
  const int  wid  = (int)((blockIdx.x * blockDim.x + threadIdx.x) >> 5);
  const int  c    = wid >> 9;                          // / (BN/32 = 512)
  const int  tile = (wid & 511) << 5;                  // column base (step 32)
  const bool hi   = lane >= 16;
  const int  mrow = lane & 15;

  // Constant triangular A (16x32 f16). A-layout (contiguous K): lane<16 elements
  // 0..15 -> K=0..15; lane>=16: 0..15 -> K=16..31.
  v16h A;
#pragma unroll
  for (int e = 0; e < 16; ++e) {
    int k = hi ? e + 16 : e;
    float av = (k < 16 && k <= mrow) ? (float)(mrow - k + 1) : 0.f;
    A[e] = (_Float16)av;
  }

  // Per-lane: own column (tile+lane) spike bits + carries entering the chunk.
  const size_t base  = (size_t)c * BN + tile + lane;
  uint32_t bits32 = spk[base];
  float    s1c    = s1_in[base];
  float    zc     = z_in[base];

  const v8f czero = {};

#pragma unroll
  for (int s = 0; s < 2; ++s) {                        // two 16-step sub-blocks
    uint32_t my = (bits32 >> (16 * s)) & 0xFFFFu;      // own column, this block
    uint32_t pb = (uint32_t)__shfl((int)my, lane | 16, 32); // partner col (+16)

    // B layout: lane<16 element e = spike[K=e][col]; lanes>=16 cover K>=16 (zero).
    v16h B1 = expand_bits_f16(hi ? 0u : my);
    v16h B2 = expand_bits_f16(hi ? 0u : pb);

    v8f d1 = __builtin_amdgcn_wmma_f32_16x16x32_f16(false, A, false, B1,
                                                    (short)0, czero, false, false);
    v8f d2 = __builtin_amdgcn_wmma_f32_16x16x32_f16(false, A, false, B2,
                                                    (short)0, czero, false, false);

    // Route per-column carries to the D layout (lane L -> col L%16 [D1] / 16+L%16 [D2]).
    float cz1 = __shfl(zc,  mrow,      32);
    float cs1 = __shfl(s1c, mrow,      32);
    float cz2 = __shfl(zc,  lane | 16, 32);
    float cs2 = __shfl(s1c, lane | 16, 32);

    const int tbase = c * LCHUNK + s * 16 + (hi ? 8 : 0);
#pragma unroll
    for (int v = 0; v < 8; ++v) {
      int   row_t = tbase + v;
      float w     = (float)((hi ? 8 : 0) + v + 1);     // (i+1)
      float z1 = d1[v] + cz1 + w * cs1;
      float z2 = d2[v] + cz2 + w * cs2;
      size_t i1 = (size_t)row_t * BN + tile + mrow;    // cols tile..tile+15
      __builtin_nontemporal_store(z1, &z_out[i1]);
      __builtin_nontemporal_store(z2, &z_out[i1 + 16]);
      __builtin_nontemporal_store((z1 == 1.0f) ? 1.0f : 0.0f, &gz_out[i1]);
      __builtin_nontemporal_store((z2 == 1.0f) ? 1.0f : 0.0f, &gz_out[i1 + 16]);
    }

    // Carry update for own column: Zl = D[15][own col].
    float zl1 = __shfl(d1[7], lane | 16, 32);          // row 15 of D1 lives in lanes 16..31
    float Zl  = hi ? d2[7] : zl1;
    float S   = (float)__popc((int)my);                // own-column spike count
    zc  = zc + 16.f * s1c + Zl;
    s1c = s1c + S;
  }
}

// -------------------------------------------------------------------------
extern "C" void kernel_launch(void* const* d_in, const int* in_sizes, int n_in,
                              void* d_out, int out_size, void* d_ws, size_t ws_size,
                              hipStream_t stream) {
  const float* cur   = (const float*)d_in[0];   // (T,B,N)
  const float* beta  = (const float*)d_in[1];   // (N,)
  const float* vinit = (const float*)d_in[2];   // (B,N)
  const float* vth   = (const float*)d_in[3];   // (T,B,N)

  float* gz_out = (float*)d_out;                       // (T,B,N)
  float* z_out  = gz_out + (size_t)T_LEN * BN;         // (T,B,N)
  float* mlast  = z_out  + (size_t)T_LEN * BN;         // (B,N)

  // Workspace: 7 arrays of CBN 32-bit words = 14 MB.
  float*    ws    = (float*)d_ws;
  float*    mend  = ws;
  float*    m_in  = ws + (size_t)1 * CBN;
  uint32_t* spk   = (uint32_t*)(ws + (size_t)2 * CBN);
  float*    Sc    = ws + (size_t)3 * CBN;
  float*    Zc    = ws + (size_t)4 * CBN;
  float*    s1_in = ws + (size_t)5 * CBN;
  float*    z_in  = ws + (size_t)6 * CBN;

  dim3 blk(256);
  pass1_mend      <<<dim3(CBN / 256), blk, 0, stream>>>(cur, beta, mend);
  pass2_mcombine  <<<dim3(BN  / 256), blk, 0, stream>>>(beta, vinit, mend, m_in, mlast);
  pass3_spikes    <<<dim3(CBN / 256), blk, 0, stream>>>(cur, vth, beta, m_in, spk, Sc, Zc);
  pass4_szcombine <<<dim3(BN  / 256), blk, 0, stream>>>(Sc, Zc, s1_in, z_in);
  pass5_wmma_out  <<<dim3(CBN / 32 / 8), blk, 0, stream>>>(spk, s1_in, z_in, gz_out, z_out);
}
